// GATv2Classifier_44744969289869
// MI455X (gfx1250) — matrix operand order repaired and checked by reading, not compiled, not run on hardware
//
#include <hip/hip_runtime.h>
#include <hip/hip_bf16.h>
#include <stdint.h>

// ---------------------------------------------------------------------------
// GATv2 classifier for MI455X (gfx1250, wave32, WMMA + async-to-LDS).
//   encoder (N,8)->(N,64) relu
//   2x GATv2: fused xl/xr WMMA GEMM (LDS-staged A via global_load_async_to_lds),
//             scatter-softmax over edges (f16 gathers, f32 math)
//   mean-pool per graph, MLP head with LayerNorm
// ---------------------------------------------------------------------------

typedef __attribute__((ext_vector_type(16))) _Float16 f16x16;
typedef __attribute__((ext_vector_type(4)))  _Float16 f16x4;
typedef __attribute__((ext_vector_type(8)))  float    f32x8;
typedef __attribute__((ext_vector_type(4)))  int      i32x4;
typedef __attribute__((ext_vector_type(8)))  int      i32x8;

#define HD 256   // H*C
#define NH 4     // heads
#define HC 64    // per-head channels
#define NGRAPH 64

// ---------------- float <-> orderable uint (for atomic segment-max) --------
__device__ __forceinline__ unsigned encf(float f) {
  unsigned u = __float_as_uint(f);
  return (u & 0x80000000u) ? ~u : (u | 0x80000000u);
}
__device__ __forceinline__ float decf(unsigned u) {
  return __uint_as_float((u & 0x80000000u) ? (u ^ 0x80000000u) : ~u);
}

// ---------------- weight pack: W[K,256] f32 -> WMMA B-fragment order -------
// Packed index: (((ks*16 + ct)*32 + lane)*16 + i)
//   lane: col = ct*16 + (lane&15), K-half = lane>>4
//   element i: k = ks*32 + (lane>>4)*16 + i
__global__ void k_pack_w(const float* __restrict__ W, _Float16* __restrict__ Wp, int K) {
  int idx = blockIdx.x * 256 + threadIdx.x;
  if (idx >= K * HD) return;
  int i    = idx & 15;
  int lane = (idx >> 4) & 31;
  int blk  = idx >> 9;          // ks*16 + ct
  int ct   = blk & 15;
  int ks   = blk >> 4;
  int col  = ct * 16 + (lane & 15);
  int k    = ks * 32 + (lane >> 4) * 16 + i;
  Wp[idx] = (_Float16)W[(size_t)k * HD + col];
}

// ---------------- encoder: h = relu(x @ Wenc + benc), write f16 ------------
__global__ void k_encoder(const float* __restrict__ x, const float* __restrict__ W,
                          const float* __restrict__ b, _Float16* __restrict__ hf16, int n) {
  int idx = blockIdx.x * 256 + threadIdx.x;
  if (idx >= n * 64) return;
  int node = idx >> 6;
  int c    = idx & 63;
  float s = b[c];
#pragma unroll
  for (int k = 0; k < 8; ++k) s += x[node * 8 + k] * W[k * 64 + c];
  hf16[idx] = (_Float16)fmaxf(s, 0.0f);
}

// ---------------- fused WMMA GEMM: YL/YR[M,256]f16 = X@WL+bL / X@WR+bR -----
// Block = 256 threads = 8 waves covering 16 rows x 256 cols.
// A-tile (16x32 f16 = 1KB) staged in LDS via global_load_async_to_lds_b32,
// double-buffered; each wave does 4 WMMAs per K-step (2 col-tiles x {L,R}).
__global__ __launch_bounds__(256) void k_wmma_gemm_dual(
    const _Float16* __restrict__ X,
    const _Float16* __restrict__ WpL, const _Float16* __restrict__ WpR,
    const float* __restrict__ biasL, const float* __restrict__ biasR,
    _Float16* __restrict__ YL, _Float16* __restrict__ YR, int M, int K) {
  __shared__ __align__(64) _Float16 sA[2][16 * 32];

  const int t       = threadIdx.x;
  const int lane    = t & 31;
  const int wave    = t >> 5;
  const int rowBase = blockIdx.x << 4;
  const int kg      = lane >> 4;     // K-half select
  const int ct0     = wave * 2;
  const int ksteps  = K >> 5;

  // async-copy mapping: thread t copies 4B (2 halves) of the 16x32 A tile
  int crow = rowBase + (t >> 4);
  crow = crow < M ? crow : (M - 1);
  const size_t cOffHalf = (size_t)crow * K + (size_t)(t & 15) * 2;

  {  // stage K-step 0 into buffer 0
    unsigned long long ga = (unsigned long long)(uintptr_t)(X + cOffHalf);
    unsigned ldsoff = (unsigned)(uintptr_t)(&sA[0][0]) + (unsigned)t * 4u;
    asm volatile("global_load_async_to_lds_b32 %0, %1, off" ::"v"(ldsoff), "v"(ga)
                 : "memory");
  }
  asm volatile("s_wait_asynccnt 0x0" ::: "memory");
  __syncthreads();

  f32x8 accL0 = {0.f, 0.f, 0.f, 0.f, 0.f, 0.f, 0.f, 0.f};
  f32x8 accL1 = accL0, accR0 = accL0, accR1 = accL0;

  for (int ks = 0; ks < ksteps; ++ks) {
    if (ks + 1 < ksteps) {  // kick off next tile's async copy (overlaps WMMAs)
      unsigned long long ga =
          (unsigned long long)(uintptr_t)(X + cOffHalf + (size_t)(ks + 1) * 32);
      unsigned ldsoff = (unsigned)(uintptr_t)(&sA[(ks + 1) & 1][0]) + (unsigned)t * 4u;
      asm volatile("global_load_async_to_lds_b32 %0, %1, off" ::"v"(ldsoff), "v"(ga)
                   : "memory");
    }

    // A fragment from LDS (16-bit A 16x32 layout): ds_load_b128 x2
    const char* abase = (const char*)&sA[ks & 1][0] + (lane & 15) * 64 + kg * 32;
    i32x8 araw;
    *(i32x4*)&araw         = *(const i32x4*)(abase);
    *(((i32x4*)&araw) + 1) = *(const i32x4*)(abase + 16);
    f16x16 a = __builtin_bit_cast(f16x16, araw);

    // B fragments: 32B coalesced loads from packed weights
    const size_t wb = (((size_t)ks * 16 + ct0) * 32 + lane) * 16;
    i32x8 bl0 = *(const i32x8*)(WpL + wb);
    i32x8 bl1 = *(const i32x8*)(WpL + wb + 512);
    i32x8 br0 = *(const i32x8*)(WpR + wb);
    i32x8 br1 = *(const i32x8*)(WpR + wb + 512);
    if (ks + 1 < ksteps) {
      const size_t wn = (((size_t)(ks + 1) * 16 + ct0) * 32 + lane) * 16;
      __builtin_prefetch((const void*)(WpL + wn), 0, 3);
      __builtin_prefetch((const void*)(WpR + wn), 0, 3);
    }

    accL0 = __builtin_amdgcn_wmma_f32_16x16x32_f16(false, a, false,
              __builtin_bit_cast(f16x16, bl0), (short)0, accL0, false, false);
    accL1 = __builtin_amdgcn_wmma_f32_16x16x32_f16(false, a, false,
              __builtin_bit_cast(f16x16, bl1), (short)0, accL1, false, false);
    accR0 = __builtin_amdgcn_wmma_f32_16x16x32_f16(false, a, false,
              __builtin_bit_cast(f16x16, br0), (short)0, accR0, false, false);
    accR1 = __builtin_amdgcn_wmma_f32_16x16x32_f16(false, a, false,
              __builtin_bit_cast(f16x16, br1), (short)0, accR1, false, false);

    asm volatile("s_wait_asynccnt 0x0" ::: "memory");
    __syncthreads();
  }

  // D layout: vgpr r, lane l -> row = (l>>4)*8 + r, col = l&15
  const int c    = lane & 15;
  const int r0   = rowBase + kg * 8;
  const int col0 = ct0 * 16 + c;
  const float bL0 = biasL[col0], bL1 = biasL[col0 + 16];
  const float bR0 = biasR[col0], bR1 = biasR[col0 + 16];
#pragma unroll
  for (int r = 0; r < 8; ++r) {
    int rr = r0 + r;
    if (rr < M) {
      size_t o = (size_t)rr * HD + col0;
      YL[o]      = (_Float16)(accL0[r] + bL0);
      YL[o + 16] = (_Float16)(accL1[r] + bL1);
      YR[o]      = (_Float16)(accR0[r] + bR0);
      YR[o + 16] = (_Float16)(accR1[r] + bR1);
    }
  }
}

// ---------------- edge pass 1: logits + atomic segment-max -----------------
__global__ void k_edge_logits(const _Float16* __restrict__ xl, const _Float16* __restrict__ xr,
                              const float* __restrict__ eattr, const int* __restrict__ src,
                              const int* __restrict__ dst, const float* __restrict__ We,
                              const float* __restrict__ att, float* __restrict__ logit,
                              unsigned* __restrict__ menc, int ecount) {
  int idx = blockIdx.x * 256 + threadIdx.x;
  if (idx >= ecount * NH) return;
  int e = idx >> 2;
  int h = idx & 3;
  const f16x4*  pl = (const f16x4*)(xl + (size_t)src[e] * HD + h * HC);
  const f16x4*  pr = (const f16x4*)(xr + (size_t)dst[e] * HD + h * HC);
  const float4* pw = (const float4*)(We + h * HC);
  const float4* pa = (const float4*)(att + h * HC);
  float ea  = eattr[e];
  float acc = 0.0f;
#pragma unroll 4
  for (int j = 0; j < 16; ++j) {
    f16x4 a = pl[j], b = pr[j];
    float4 w = pw[j], at = pa[j];
    float s;
    s = (float)a[0] + (float)b[0] + ea * w.x; s = s > 0.f ? s : 0.2f * s; acc += s * at.x;
    s = (float)a[1] + (float)b[1] + ea * w.y; s = s > 0.f ? s : 0.2f * s; acc += s * at.y;
    s = (float)a[2] + (float)b[2] + ea * w.z; s = s > 0.f ? s : 0.2f * s; acc += s * at.z;
    s = (float)a[3] + (float)b[3] + ea * w.w; s = s > 0.f ? s : 0.2f * s; acc += s * at.w;
  }
  logit[idx] = acc;
  atomicMax(&menc[(size_t)dst[e] * NH + h], encf(acc));
}

// ---------------- edge pass 2: exp + atomic denominator --------------------
__global__ void k_edge_exp(const float* __restrict__ logit, const int* __restrict__ dst,
                           const unsigned* __restrict__ menc, float* __restrict__ den,
                           float* __restrict__ ex, int ecount) {
  int idx = blockIdx.x * 256 + threadIdx.x;
  if (idx >= ecount * NH) return;
  int e = idx >> 2;
  int h = idx & 3;
  float m = decf(menc[(size_t)dst[e] * NH + h]);
  float v = __expf(logit[idx] - m);
  ex[idx] = v;
  atomicAdd(&den[(size_t)dst[e] * NH + h], v);
}

// ---------------- edge pass 2.5: alpha = ex / den[dst] ---------------------
__global__ void k_alpha(float* __restrict__ ex, const int* __restrict__ dst,
                        const float* __restrict__ den, int ecount) {
  int idx = blockIdx.x * 256 + threadIdx.x;
  if (idx >= ecount * NH) return;
  int e = idx >> 2;
  int h = idx & 3;
  ex[idx] = ex[idx] / (den[(size_t)dst[e] * NH + h] + 1e-16f);
}

// ---------------- edge pass 3: scatter alpha * xl[src] ---------------------
// One thread per (edge, head, 4-chunk): 4 float atomics each.
__global__ void k_scatter(const _Float16* __restrict__ xl, const int* __restrict__ src,
                          const int* __restrict__ dst, const float* __restrict__ alpha,
                          float* __restrict__ accum, int ecount) {
  int idx = blockIdx.x * 256 + threadIdx.x;
  if (idx >= ecount * NH * 16) return;
  int e   = idx >> 6;
  int rem = idx & 63;
  int h   = rem >> 4;
  int j   = rem & 15;
  float w = alpha[(size_t)e * NH + h];
  f16x4 a = ((const f16x4*)(xl + (size_t)src[e] * HD + h * HC))[j];
  float* o = accum + (size_t)dst[e] * HD + h * HC + j * 4;
  atomicAdd(o + 0, w * (float)a[0]);
  atomicAdd(o + 1, w * (float)a[1]);
  atomicAdd(o + 2, w * (float)a[2]);
  atomicAdd(o + 3, w * (float)a[3]);
}

// ---------------- finalize: +bias, relu, write f32 + f16 -------------------
__global__ void k_finalize(float* __restrict__ accum, const float* __restrict__ bias,
                           _Float16* __restrict__ hf16, int n) {
  int idx = blockIdx.x * 256 + threadIdx.x;
  if (idx >= n * HD) return;
  float v = fmaxf(accum[idx] + bias[idx & 255], 0.0f);
  accum[idx] = v;
  hf16[idx]  = (_Float16)v;
}

// ---------------- per-graph mean pool (atomics) ----------------------------
__global__ void k_pool(const float* __restrict__ h, const int* __restrict__ batch,
                       float* __restrict__ pool, float* __restrict__ cnt, int n) {
  int idx = blockIdx.x * 256 + threadIdx.x;
  if (idx >= n * HD) return;
  int node = idx >> 8;
  int c    = idx & 255;
  int g    = batch[node];
  atomicAdd(&pool[(size_t)g * HD + c], h[idx]);
  if (c == 0) atomicAdd(&cnt[g], 1.0f);
}

// ---------------- MLP head: 256 -> 128 -> LN -> relu -> 64 -> relu -> 1 ----
__global__ __launch_bounds__(128) void k_head(
    const float* __restrict__ pool, const float* __restrict__ cnt,
    const float* __restrict__ Wp1, const float* __restrict__ bp1,
    const float* __restrict__ lng, const float* __restrict__ lnb,
    const float* __restrict__ Wp2, const float* __restrict__ bp2,
    const float* __restrict__ Wh, const float* __restrict__ bh,
    float* __restrict__ out) {
  __shared__ float p[HD];
  __shared__ float q[128];
  __shared__ float r2[64];
  __shared__ float stats[2];
  const int g = blockIdx.x;
  const int t = threadIdx.x;
  float inv = 1.0f / fmaxf(cnt[g], 1.0f);
  for (int c = t; c < HD; c += 128) p[c] = pool[(size_t)g * HD + c] * inv;
  __syncthreads();
  float s = bp1[t];
  for (int c = 0; c < HD; ++c) s += p[c] * Wp1[c * 128 + t];
  q[t] = s;
  __syncthreads();
  if (t == 0) {
    float mu = 0.f;
    for (int j = 0; j < 128; ++j) mu += q[j];
    mu *= (1.0f / 128.0f);
    float var = 0.f;
    for (int j = 0; j < 128; ++j) { float d = q[j] - mu; var += d * d; }
    stats[0] = mu;
    stats[1] = var * (1.0f / 128.0f);
  }
  __syncthreads();
  float qn = (q[t] - stats[0]) * rsqrtf(stats[1] + 1e-5f) * lng[t] + lnb[t];
  __syncthreads();
  q[t] = fmaxf(qn, 0.0f);
  __syncthreads();
  if (t < 64) {
    float s2 = bp2[t];
    for (int c = 0; c < 128; ++c) s2 += q[c] * Wp2[c * 64 + t];
    r2[t] = fmaxf(s2, 0.0f);
  }
  __syncthreads();
  if (t == 0) {
    float o = bh[0];
    for (int c = 0; c < 64; ++c) o += r2[c] * Wh[c];
    out[g] = o;
  }
}

// ---------------------------------------------------------------------------
extern "C" void kernel_launch(void* const* d_in, const int* in_sizes, int n_in,
                              void* d_out, int out_size, void* d_ws, size_t ws_size,
                              hipStream_t stream) {
  (void)n_in; (void)out_size; (void)ws_size;

  const float* x     = (const float*)d_in[0];
  const float* eattr = (const float*)d_in[1];
  const int*   src   = (const int*)d_in[2];
  const int*   dst   = (const int*)d_in[3];
  const int*   batch = (const int*)d_in[4];
  const float* Wenc  = (const float*)d_in[5];
  const float* benc  = (const float*)d_in[6];
  const float* gWl[2]   = {(const float*)d_in[7],  (const float*)d_in[14]};
  const float* gbl[2]   = {(const float*)d_in[8],  (const float*)d_in[15]};
  const float* gWr[2]   = {(const float*)d_in[9],  (const float*)d_in[16]};
  const float* gbr[2]   = {(const float*)d_in[10], (const float*)d_in[17]};
  const float* gWe[2]   = {(const float*)d_in[11], (const float*)d_in[18]};
  const float* gatt[2]  = {(const float*)d_in[12], (const float*)d_in[19]};
  const float* gbias[2] = {(const float*)d_in[13], (const float*)d_in[20]};
  const float* Wp1 = (const float*)d_in[21];
  const float* bp1 = (const float*)d_in[22];
  const float* lng = (const float*)d_in[23];
  const float* lnb = (const float*)d_in[24];
  const float* Wp2 = (const float*)d_in[25];
  const float* bp2 = (const float*)d_in[26];
  const float* Wh  = (const float*)d_in[27];
  const float* bh  = (const float*)d_in[28];

  const int N = in_sizes[0] / 8;   // nodes
  const int E = in_sizes[2];       // edges

  // ---- workspace sub-allocation (256B aligned) ----
  char*  ws  = (char*)d_ws;
  size_t off = 0;
  auto suballoc = [&](size_t bytes) -> void* {
    void* p = ws + off;
    off += (bytes + 255) & ~(size_t)255;
    return p;
  };
  _Float16* hf16  = (_Float16*)suballoc((size_t)N * HD * sizeof(_Float16));
  _Float16* xl    = (_Float16*)suballoc((size_t)N * HD * sizeof(_Float16));
  _Float16* xr    = (_Float16*)suballoc((size_t)N * HD * sizeof(_Float16));
  float*    accum = (float*)   suballoc((size_t)N * HD * sizeof(float));
  float*    logit = (float*)   suballoc((size_t)E * NH * sizeof(float));
  float*    exbuf = (float*)   suballoc((size_t)E * NH * sizeof(float));
  unsigned* menc  = (unsigned*)suballoc((size_t)N * NH * sizeof(unsigned));
  float*    den   = (float*)   suballoc((size_t)N * NH * sizeof(float));
  _Float16* wpa   = (_Float16*)suballoc((size_t)HD * HD * sizeof(_Float16));
  _Float16* wpb   = (_Float16*)suballoc((size_t)HD * HD * sizeof(_Float16));
  float*    pool  = (float*)   suballoc((size_t)NGRAPH * HD * sizeof(float));
  float*    cnt   = (float*)   suballoc((size_t)NGRAPH * sizeof(float));

  const int T = 256;
  // encoder
  k_encoder<<<(N * 64 + T - 1) / T, T, 0, stream>>>(x, Wenc, benc, hf16, N);

  for (int L = 0; L < 2; ++L) {
    const int K = (L == 0) ? 64 : HD;
    // pack weights into WMMA B-fragment order
    k_pack_w<<<(K * HD + T - 1) / T, T, 0, stream>>>(gWl[L], wpa, K);
    k_pack_w<<<(K * HD + T - 1) / T, T, 0, stream>>>(gWr[L], wpb, K);
    // xl = h@Wl+bl ; xr = h@Wr+br  (fused WMMA GEMM, f16 outputs)
    const int mblocks = (N + 15) / 16;
    k_wmma_gemm_dual<<<mblocks, T, 0, stream>>>(hf16, wpa, wpb, gbl[L], gbr[L], xl, xr, N, K);
    // scatter-softmax
    hipMemsetAsync(menc, 0, (size_t)N * NH * sizeof(unsigned), stream);
    hipMemsetAsync(den, 0, (size_t)N * NH * sizeof(float), stream);
    hipMemsetAsync(accum, 0, (size_t)N * HD * sizeof(float), stream);
    k_edge_logits<<<(E * NH + T - 1) / T, T, 0, stream>>>(xl, xr, eattr, src, dst,
                                                          gWe[L], gatt[L], logit, menc, E);
    k_edge_exp<<<(E * NH + T - 1) / T, T, 0, stream>>>(logit, dst, menc, den, exbuf, E);
    k_alpha<<<(E * NH + T - 1) / T, T, 0, stream>>>(exbuf, dst, den, E);
    k_scatter<<<((size_t)E * NH * 16 + T - 1) / T, T, 0, stream>>>(xl, src, dst, exbuf,
                                                                   accum, E);
    k_finalize<<<((size_t)N * HD + T - 1) / T, T, 0, stream>>>(accum, gbias[L], hf16, N);
  }

  // mean pool + head
  hipMemsetAsync(pool, 0, (size_t)NGRAPH * HD * sizeof(float), stream);
  hipMemsetAsync(cnt, 0, (size_t)NGRAPH * sizeof(float), stream);
  k_pool<<<((size_t)N * HD + T - 1) / T, T, 0, stream>>>(accum, batch, pool, cnt, N);
  k_head<<<NGRAPH, 128, 0, stream>>>(pool, cnt, Wp1, bp1, lng, lnb, Wp2, bp2, Wh, bh,
                                     (float*)d_out);
}
